// NeuronNemotronRouter_25890062860803
// MI455X (gfx1250) — matrix operand order repaired and checked by reading, not compiled, not run on hardware
//
#include <hip/hip_runtime.h>
#include <hip/hip_bf16.h>

// MoE router: logits = X @ W^T, sigmoid, +bias top-8, normalized weights.
// T=16384, H=7168, E=256, TOPK=8. bf16 WMMA w/ fp32 accumulate (memory-bound
// balanced design for MI455X: 478MB @ 23.3TB/s ~= 20us ~= 60GFLOP at bf16 rate).

typedef __bf16 bf16;
typedef __attribute__((ext_vector_type(16))) __bf16 v16bf;
typedef __attribute__((ext_vector_type(8)))  __bf16 bf16x8;
typedef __attribute__((ext_vector_type(4)))  __bf16 bf16x4;
typedef __attribute__((ext_vector_type(8)))  float  v8f;
typedef __attribute__((ext_vector_type(4)))  float  f32x4;

#define T_TOK 16384
#define HID   7168
#define NEXP  256
#define TOPK  8
#define BM    64
#define BK    64
#define LDSTR 72      // bf16 elements per LDS row (144B: 16B-aligned, conflict-free)
#define NKT   (HID / BK)   // 112 stages

struct Phase1 {
    bf16 A[2][BM][LDSTR];       // token tile,  double buffered
    bf16 W[2][NEXP][LDSTR];     // expert tile, double buffered
};
struct Phase2 {
    float S[BM][257];           // raw sigmoid scores, padded stride (bank-conflict-free scans)
    float cK[BM][4][TOPK];      // per-quarter candidate corrected scores (sorted desc)
    float cV[BM][4][TOPK];      // per-quarter candidate raw scores
    int   cI[BM][4][TOPK];      // per-quarter candidate expert indices
};
union Smem { Phase1 p1; Phase2 p2; };

union FragU { v16bf v; bf16x8 h[2]; };

__device__ __forceinline__ v8f wmma_bf16(v16bf a, v16bf b, v8f c) {
    return __builtin_amdgcn_wmma_f32_16x16x32_bf16(
        /*neg_a=*/false, a, /*neg_b=*/false, b,
        /*c_mod=*/(short)0, c, /*reuse_a=*/false, /*reuse_b=*/false);
}

__device__ __forceinline__ bf16x4 cvt4(f32x4 v) {
    bf16x4 r;
    r.x = (bf16)v.x; r.y = (bf16)v.y; r.z = (bf16)v.z; r.w = (bf16)v.w;
    return r;
}

__global__ __launch_bounds__(256)
void router_kernel(const float* __restrict__ X,
                   const float* __restrict__ Wm,
                   const float* __restrict__ bias,
                   int*   __restrict__ outIdx,
                   float* __restrict__ outW) {
    __shared__ Smem  sm;
    __shared__ float sBias[NEXP];

    const int tid  = threadIdx.x;
    const int lane = tid & 31;
    const int wave = tid >> 5;
    const int m_pair = wave >> 2;   // 0..1 -> M blocks {2*m_pair, 2*m_pair+1}
    const int n_quad = wave & 3;    // 0..3 -> N tiles  {4*n_quad .. 4*n_quad+3}
    const int bm0 = blockIdx.x * BM;

    sBias[tid] = bias[tid];         // blockDim == NEXP == 256

    // ---- global load mapping: 4 lanes per row, 64B contiguous per lane-quad ----
    const int lr = tid >> 2;        // row group 0..63
    const int lq = tid & 3;         // quarter within row

    v8f acc[2][4];
    const v8f vzero = {0.f,0.f,0.f,0.f,0.f,0.f,0.f,0.f};
#pragma unroll
    for (int mi = 0; mi < 2; ++mi)
#pragma unroll
        for (int ni = 0; ni < 4; ++ni) acc[mi][ni] = vzero;

    // ---- fragment loaders (ISA 7.12.2 layouts; D = A * W^T, both row-major) ----
    auto ldA = [&](int buf, int mblk, int ko) -> v16bf {
        const int m = lane & 15, h = lane >> 4;
        const bf16* p = &sm.p1.A[buf][mblk * 16 + m][ko + h * 8];
        FragU u;
        u.h[0] = *(const bf16x8*)p;          // K = ko + h*8 .. +7
        u.h[1] = *(const bf16x8*)(p + 16);   // K = ko + 16 + h*8 .. +7
        return u.v;
    };
    auto ldB = [&](int buf, int ntile, int ko) -> v16bf {
        const int n = lane & 15, kh = lane >> 4;
        const bf16* p = &sm.p1.W[buf][ntile * 16 + n][ko + kh * 8];
        FragU u;
        u.h[0] = *(const bf16x8*)p;          // K = ko + kh*8 .. +7
        u.h[1] = *(const bf16x8*)(p + 16);   // K = ko + 16 + kh*8 .. +7
        return u.v;
    };

    auto loadRegs = [&](int k0, f32x4* rA, f32x4* rW) {
        const float* xa = X + (size_t)(bm0 + lr) * HID + k0 + lq * 4;
#pragma unroll
        for (int j = 0; j < 4; ++j) rA[j] = *(const f32x4*)(xa + j * 16);
#pragma unroll
        for (int c = 0; c < 4; ++c) {
            const float* wr = Wm + (size_t)(lr + c * 64) * HID + k0 + lq * 4;
#pragma unroll
            for (int j = 0; j < 4; ++j) rW[c * 4 + j] = *(const f32x4*)(wr + j * 16);
        }
    };
    auto storeTile = [&](int buf, const f32x4* rA, const f32x4* rW) {
#pragma unroll
        for (int j = 0; j < 4; ++j)
            *(bf16x4*)&sm.p1.A[buf][lr][lq * 4 + j * 16] = cvt4(rA[j]);
#pragma unroll
        for (int c = 0; c < 4; ++c)
#pragma unroll
            for (int j = 0; j < 4; ++j)
                *(bf16x4*)&sm.p1.W[buf][lr + c * 64][lq * 4 + j * 16] = cvt4(rW[c * 4 + j]);
    };

    // ---- prologue: stage 0 ----
    {
        f32x4 rA[4], rW[16];
        loadRegs(0, rA, rW);
        storeTile(0, rA, rW);
    }
    __syncthreads();

    // ---- main K loop, double buffered ----
    for (int kt = 0; kt < NKT; ++kt) {
        const int buf = kt & 1;
        f32x4 rA[4], rW[16];
        const bool pre = (kt + 1) < NKT;
        if (pre) loadRegs((kt + 1) * BK, rA, rW);   // issue HBM loads early

#pragma unroll
        for (int ks = 0; ks < 2; ++ks) {            // two K=32 steps per stage
            const int ko = ks * 32;
            v16bf a0 = ldA(buf, m_pair * 2 + 0, ko);
            v16bf a1 = ldA(buf, m_pair * 2 + 1, ko);
#pragma unroll
            for (int ni = 0; ni < 4; ++ni) {
                v16bf b = ldB(buf, n_quad * 4 + ni, ko);
                acc[0][ni] = wmma_bf16(a0, b, acc[0][ni]);
                acc[1][ni] = wmma_bf16(a1, b, acc[1][ni]);
            }
        }
        if (pre) storeTile(buf ^ 1, rA, rW);
        __syncthreads();
    }

    // ---- phase 2a: sigmoid + scatter raw scores to LDS (union re-use is safe post-barrier) ----
#pragma unroll
    for (int mi = 0; mi < 2; ++mi)
#pragma unroll
        for (int ni = 0; ni < 4; ++ni)
#pragma unroll
            for (int g = 0; g < 8; ++g) {
                const int r = (m_pair * 2 + mi) * 16 + (lane >> 4) * 8 + g;  // C layout: M
                const int c = (n_quad * 4 + ni) * 16 + (lane & 15);          // C layout: N
                const float x = acc[mi][ni][g];
                sm.p2.S[r][c] = 1.0f / (1.0f + __expf(-x));
            }
    __syncthreads();

    // ---- phase 2b: 4 threads per token, each scans 64 experts, branchless top-8 ----
    {
        const int tk = tid >> 2, grp = tid & 3;
        float key[TOPK], val[TOPK];
        int   idx[TOPK];
#pragma unroll
        for (int j = 0; j < TOPK; ++j) { key[j] = -3.0e38f; val[j] = 0.f; idx[j] = 0; }

        for (int e0 = 0; e0 < 64; ++e0) {
            const int e = grp * 64 + e0;
            const float s  = sm.p2.S[tk][e];
            float nk = s + sBias[e];     // select on corrected score
            float nv = s;                // weight from raw score
            int   ne = e;
#pragma unroll
            for (int j = 0; j < TOPK; ++j) {   // strict > : lowest index wins ties (JAX top_k)
                const bool sw = nk > key[j];
                const float tkey = key[j], tval = val[j]; const int tix = idx[j];
                key[j] = sw ? nk : tkey; val[j] = sw ? nv : tval; idx[j] = sw ? ne : tix;
                nk = sw ? tkey : nk;     nv = sw ? tval : nv;     ne = sw ? tix : ne;
            }
        }
#pragma unroll
        for (int j = 0; j < TOPK; ++j) {
            sm.p2.cK[tk][grp][j] = key[j];
            sm.p2.cV[tk][grp][j] = val[j];
            sm.p2.cI[tk][grp][j] = idx[j];
        }
    }
    __syncthreads();

    // ---- phase 2c: merge 4 sorted candidate lists per token, normalize, write out ----
    if (tid < BM) {
        int p[4] = {0, 0, 0, 0};
        float ow[TOPK]; int oi[TOPK];
        float wsum = 0.f;
#pragma unroll
        for (int k = 0; k < TOPK; ++k) {
            int bg = 0; float bk = -3.4e38f;
#pragma unroll
            for (int g = 0; g < 4; ++g) {   // scan low->high group: lowest expert idx wins ties
                const float kk = sm.p2.cK[tid][g][p[g]];
                if (kk > bk) { bk = kk; bg = g; }
            }
            ow[k] = sm.p2.cV[tid][bg][p[bg]];
            oi[k] = sm.p2.cI[tid][bg][p[bg]];
            p[bg]++;
            wsum += ow[k];
        }
        const float scale = 2.5f / (wsum + 1e-20f);
        const size_t tok = (size_t)bm0 + tid;
#pragma unroll
        for (int k = 0; k < TOPK; ++k) {
            outIdx[tok * TOPK + k] = oi[k];
            outW[tok * TOPK + k]   = ow[k] * scale;
        }
    }
}

extern "C" void kernel_launch(void* const* d_in, const int* in_sizes, int n_in,
                              void* d_out, int out_size, void* d_ws, size_t ws_size,
                              hipStream_t stream) {
    const float* X    = (const float*)d_in[0];   // (16384, 7168) fp32
    const float* Wm   = (const float*)d_in[1];   // (256, 7168)   fp32
    const float* bias = (const float*)d_in[2];   // (256,)        fp32
    int*   outIdx = (int*)d_out;                          // (T, 8) int32, first in tuple
    float* outW   = (float*)d_out + (size_t)T_TOK * TOPK; // (T, 8) fp32, second

    dim3 grid(T_TOK / BM), block(256);
    router_kernel<<<grid, block, 0, stream>>>(X, Wm, bias, outIdx, outW);
}
